// PositionMultiheadCrossAttentionWeights_21784074126022
// MI455X (gfx1250) — matrix operand order repaired and checked by reading, not compiled, not run on hardware
//
#include <hip/hip_runtime.h>
#include <hip/hip_bf16.h>
#include <stdint.h>

// ---------------------------------------------------------------------------
// Problem constants (fixed by the reference)
// ---------------------------------------------------------------------------
#define S        2048          // sequence length (q_len == k_len)
#define BATCH    4
#define HEADS    8
#define QHD      32            // head dim
#define EMBED    512
#define IN_PROJ  544
#define MASK_VAL (-1000.0f)

#define NTILES   (S / 16)      // 128 16-row tiles along sequence
// fragment buffer: [b][h][tile][512 halfs]  (32 lanes * 16 f16 each)
#define FRAG_TILES   ((size_t)BATCH * HEADS * NTILES)
#define FRAG_ELEMS   (FRAG_TILES * 512)

typedef __attribute__((ext_vector_type(16))) _Float16 v16h;
typedef __attribute__((ext_vector_type(8)))  float    v8f;

// ---------------------------------------------------------------------------
// Sinusoidal PE value: pe[pos][e]
// ---------------------------------------------------------------------------
__device__ __forceinline__ float pe_val(int pos, int e) {
    const float factor = -9.210340371976184f / 512.0f; // -ln(10000)/EMBED
    float div = __expf((float)(e & ~1) * factor);
    float ang = (float)pos * div;
    return (e & 1) ? __cosf(ang) : __sinf(ang);
}

// ---------------------------------------------------------------------------
// Projection kernel:  q = (query + PE) @ Wq[:, :256] + bq   (z == 0)
//                     k =  key        @ Wk[:, 256:512] + bk (z == 1)
// Output written as f16 WMMA fragments:
//   A-layout (q): lane = m + 16*((d>>3)&1), slot = (d&7) + 8*(d>=16)
//   B-layout (k): lane = n + 16*((d>>3)&1), slot = (d&7) + 8*(d>=16)
// Grid: (128 row-tiles of 64 flat rows, 4 col-groups of 64, 2 proj), 256 thr.
// ---------------------------------------------------------------------------
__global__ __launch_bounds__(256)
void proj_kernel(const float* __restrict__ query, const float* __restrict__ key,
                 const float* __restrict__ Wq, const float* __restrict__ bq,
                 const float* __restrict__ Wk, const float* __restrict__ bk,
                 _Float16* __restrict__ qfrag, _Float16* __restrict__ kfrag)
{
    const int isK    = blockIdx.z;
    const int tile   = blockIdx.x;              // covers q-tile `tile` (16 q, 4 b)
    const int cg64   = blockIdx.y;              // 64-column group inside the 256 cols
    const float* in   = isK ? key : query;
    const float* W    = isK ? Wk  : Wq;
    const float* bias = isK ? bk  : bq;
    const int colBase = isK * 256 + cg64 * 64;  // absolute column in IN_PROJ

    __shared__ float Alds[64][33];              // 64 rows x 32 (padded)
    __shared__ float Wlds[32][64];              // 32 e   x 64 cols

    const int r  = threadIdx.x & 63;            // flat row within block (= q*4+b local)
    const int cg = threadIdx.x >> 6;            // 16-col subgroup (0..3)

    float acc[16];
    #pragma unroll
    for (int i = 0; i < 16; ++i) acc[i] = 0.0f;

    for (int kc = 0; kc < 16; ++kc) {           // 512 / 32 K-chunks
        const int e0 = kc * 32;
        // ---- stage A chunk (64 x 32 floats) ----
        #pragma unroll
        for (int u = 0; u < 2; ++u) {
            int f4 = threadIdx.x * 2 + u;       // 0..511 float4's
            int rr = f4 >> 3;
            int ee = (f4 & 7) * 4;
            const float4 v = *(const float4*)(in + ((size_t)(tile * 64 + rr)) * EMBED + e0 + ee);
            float vx = v.x, vy = v.y, vz = v.z, vw = v.w;
            if (!isK) {
                int pos = tile * 16 + (rr >> 2);
                vx += pe_val(pos, e0 + ee + 0);
                vy += pe_val(pos, e0 + ee + 1);
                vz += pe_val(pos, e0 + ee + 2);
                vw += pe_val(pos, e0 + ee + 3);
            }
            Alds[rr][ee + 0] = vx; Alds[rr][ee + 1] = vy;
            Alds[rr][ee + 2] = vz; Alds[rr][ee + 3] = vw;
        }
        // ---- stage W chunk (32 x 64 floats) ----
        #pragma unroll
        for (int u = 0; u < 2; ++u) {
            int f4 = threadIdx.x * 2 + u;
            int ee = f4 >> 4;
            int cc = (f4 & 15) * 4;
            const float4 v = *(const float4*)(W + ((size_t)(e0 + ee)) * IN_PROJ + colBase + cc);
            Wlds[ee][cc + 0] = v.x; Wlds[ee][cc + 1] = v.y;
            Wlds[ee][cc + 2] = v.z; Wlds[ee][cc + 3] = v.w;
        }
        __syncthreads();
        #pragma unroll 4
        for (int e = 0; e < 32; ++e) {
            const float av = Alds[r][e];
            #pragma unroll
            for (int i = 0; i < 16; ++i)
                acc[i] = fmaf(av, Wlds[e][cg * 16 + i], acc[i]);
        }
        __syncthreads();
    }

    // ---- epilogue: bias + fragment-layout store ----
    const int b = r & 3;                        // rows are (q*BATCH + b)
    const int m = r >> 2;                       // row within 16-q tile
    _Float16* frag = isK ? kfrag : qfrag;
    #pragma unroll
    for (int i = 0; i < 16; ++i) {
        const int c  = cg64 * 64 + cg * 16 + i; // 0..255 local projected column
        const int h  = c >> 5;
        const int d  = c & 31;
        const float val = acc[i] + bias[isK * 256 + c];
        const size_t base = ((size_t)((b * HEADS + h) * NTILES + tile)) * 512;
        int lane_, slot_;
        if (!isK) { lane_ = m + 16 * ((d >> 3) & 1); slot_ = (d & 7) + ((d >= 16) ? 8 : 0); }
        else      { lane_ = m + 16 * ((d >> 3) & 1); slot_ = (d & 7) + ((d >= 16) ? 8 : 0); }
        frag[base + lane_ * 16 + slot_] = (_Float16)val;
    }
}

// ---------------------------------------------------------------------------
// Fused scores + mask + softmax kernel.
// Grid: (128 q-tiles, BATCH, HEADS), 256 threads = 8 waves (wave32).
// Each block computes a 16 x 2048 stripe of scores with 128 WMMAs
// (16 per wave), keeps everything in VGPRs, then softmaxes and writes once.
// ---------------------------------------------------------------------------
__global__ __launch_bounds__(256)
void scores_softmax_kernel(const _Float16* __restrict__ qfrag,
                           const _Float16* __restrict__ kfrag,
                           const unsigned char* __restrict__ kpm,   // (B,S) bool
                           const unsigned char* __restrict__ amask, // (S,S) bool
                           float* __restrict__ out)                  // (H,B,S,S)
{
    const int qt   = blockIdx.x;
    const int b    = blockIdx.y;
    const int h    = blockIdx.z;
    const int lane = threadIdx.x & 31;
    const int wave = threadIdx.x >> 5;
    const int hi   = lane >> 4;      // 0: rows 0-7, 1: rows 8-15 (C layout)
    const int nn   = lane & 15;      // column within 16-wide tile

    __shared__ float redmax[8][16];
    __shared__ float redsum[8][16];

    // A fragment for this q-tile (shared by all 16 WMMAs of this wave)
    const _Float16* qbase = qfrag + ((size_t)((b * HEADS + h) * NTILES + qt)) * 512;
    const v16h a = *(const v16h*)(qbase + lane * 16);

    const _Float16* kball = kfrag + ((size_t)(b * HEADS + h)) * NTILES * 512;

    // ---- 16 WMMAs: K=32 head dim handled in a single instruction each ----
    v8f acc[16];
    #pragma unroll
    for (int t = 0; t < 16; ++t) {
        const int kt = wave + t * 8;
        const v16h bf = *(const v16h*)(kball + (size_t)kt * 512 + lane * 16);
        v8f c = {0.f, 0.f, 0.f, 0.f, 0.f, 0.f, 0.f, 0.f};
        acc[t] = __builtin_amdgcn_wmma_f32_16x16x32_f16(
            /*neg_a=*/false, a, /*neg_b=*/false, bf,
            /*c_mod=*/(short)0, c, /*reuse_a=*/false, /*reuse_b=*/false);
    }

    // ---- masking + per-lane row max ----
    float pm[8];
    #pragma unroll
    for (int j = 0; j < 8; ++j) pm[j] = -3.0e38f;

    #pragma unroll
    for (int t = 0; t < 16; ++t) {
        const int kt = wave + t * 8;
        const int k0 = kt * 16 + nn;
        const unsigned int kmb = kpm[(size_t)b * S + k0];
        #pragma unroll
        for (int j = 0; j < 8; ++j) {
            const int M  = j + 8 * hi;
            const int qq = qt * 16 + M;
            const unsigned int mb = kmb | amask[(size_t)qq * S + k0];
            float s = acc[t][j];
            if (mb) s = MASK_VAL;
            acc[t][j] = s;
            pm[j] = fmaxf(pm[j], s);
        }
    }

    // reduce max across the 16 lanes holding the same rows
    #pragma unroll
    for (int j = 0; j < 8; ++j) {
        #pragma unroll
        for (int sft = 1; sft < 16; sft <<= 1)
            pm[j] = fmaxf(pm[j], __shfl_xor(pm[j], sft, 32));
    }
    if (nn == 0) {
        #pragma unroll
        for (int j = 0; j < 8; ++j) redmax[wave][j + 8 * hi] = pm[j];
    }
    __syncthreads();

    float m[8];
    #pragma unroll
    for (int j = 0; j < 8; ++j) {
        float v = redmax[0][j + 8 * hi];
        #pragma unroll
        for (int w = 1; w < 8; ++w) v = fmaxf(v, redmax[w][j + 8 * hi]);
        m[j] = v;
    }

    // ---- exp in place + per-lane row sum ----
    float pl[8];
    #pragma unroll
    for (int j = 0; j < 8; ++j) pl[j] = 0.0f;
    #pragma unroll
    for (int t = 0; t < 16; ++t) {
        #pragma unroll
        for (int j = 0; j < 8; ++j) {
            const float e = __expf(acc[t][j] - m[j]);
            acc[t][j] = e;
            pl[j] += e;
        }
    }
    #pragma unroll
    for (int j = 0; j < 8; ++j) {
        #pragma unroll
        for (int sft = 1; sft < 16; sft <<= 1)
            pl[j] += __shfl_xor(pl[j], sft, 32);
    }
    if (nn == 0) {
        #pragma unroll
        for (int j = 0; j < 8; ++j) redsum[wave][j + 8 * hi] = pl[j];
    }
    __syncthreads();

    float inv[8];
    #pragma unroll
    for (int j = 0; j < 8; ++j) {
        float v = redsum[0][j + 8 * hi];
        #pragma unroll
        for (int w = 1; w < 8; ++w) v += redsum[w][j + 8 * hi];
        inv[j] = 1.0f / v;
    }

    // ---- single write of final probabilities ----
    const size_t obase = (((size_t)(h * BATCH + b)) * S + (size_t)qt * 16) * S;
    #pragma unroll
    for (int t = 0; t < 16; ++t) {
        const int kt = wave + t * 8;
        const int k0 = kt * 16 + nn;
        #pragma unroll
        for (int j = 0; j < 8; ++j) {
            const int M = j + 8 * hi;
            out[obase + (size_t)M * S + k0] = acc[t][j] * inv[j];
        }
    }
}

// ---------------------------------------------------------------------------
extern "C" void kernel_launch(void* const* d_in, const int* in_sizes, int n_in,
                              void* d_out, int out_size, void* d_ws, size_t ws_size,
                              hipStream_t stream)
{
    (void)in_sizes; (void)n_in; (void)out_size; (void)ws_size;

    const float* query = (const float*)d_in[0];
    const float* key   = (const float*)d_in[1];
    const float* Wq    = (const float*)d_in[2];
    const float* bq    = (const float*)d_in[3];
    const float* Wk    = (const float*)d_in[4];
    const float* bk    = (const float*)d_in[5];
    const unsigned char* kpm = (const unsigned char*)d_in[6]; // bool (B,S)
    const unsigned char* am  = (const unsigned char*)d_in[7]; // bool (S,S)
    float* out = (float*)d_out;

    _Float16* qfrag = (_Float16*)d_ws;              // 4 MB
    _Float16* kfrag = qfrag + FRAG_ELEMS;           // 4 MB  (ws >= 8 MB)

    proj_kernel<<<dim3(NTILES, 4, 2), 256, 0, stream>>>(
        query, key, Wq, bq, Wk, bk, qfrag, kfrag);

    scores_softmax_kernel<<<dim3(NTILES, BATCH, HEADS), 256, 0, stream>>>(
        qfrag, kfrag, kpm, am, out);
}
